// GNNClassifier_88648124990138
// MI455X (gfx1250) — matrix operand order repaired and checked by reading, not compiled, not run on hardware
//
#include <hip/hip_runtime.h>

// ---------------------------------------------------------------------------
// GNN SAGE classifier for MI455X (gfx1250, wave32, WMMA).
//
// Cost model: edge gather/scatter ~1GB HBM traffic (~45us @ 23.3TB/s) is the
// bound; dense 64->64 linears (3.3 GFLOP) are pushed onto v_wmma_f32_16x16x32_f16
// so they are noise. agg/gsum atomics are f32 adds that stay in the 192MB L2.
// ---------------------------------------------------------------------------

#define N_NODES   100000
#define N_EDGES   1000000
#define VOCAB     128
#define EMB       64
#define HID       64
#define N_CLASSES 2
#define N_GRAPHS  2048

typedef __attribute__((ext_vector_type(16))) _Float16 v16h;
typedef __attribute__((ext_vector_type(8)))  float    v8f;

// ---------------------------------------------------------------- utilities
__global__ void zero_f32(float* __restrict__ p, int n) {
  int i = blockIdx.x * blockDim.x + threadIdx.x;
  if (i < n) p[i] = 0.0f;
}

// h0[node] = emb[x[node]] ; one thread per (node, float4 chunk)
__global__ void embed_kernel(const int* __restrict__ x,
                             const float* __restrict__ emb,
                             float* __restrict__ h) {
  int t = blockIdx.x * blockDim.x + threadIdx.x;
  if (t >= N_NODES * 16) return;
  int node = t >> 4;
  int c    = (t & 15) << 2;
  int tok  = x[node];
  const float4 v = *(const float4*)(emb + tok * EMB + c);
  *(float4*)(h + node * EMB + c) = v;
}

// cnt[dst[e]] += 1  (degree, shared by both conv layers)
__global__ void degree_kernel(const int* __restrict__ dst,
                              float* __restrict__ cnt) {
  int e = blockIdx.x * blockDim.x + threadIdx.x;
  if (e < N_EDGES) atomicAdd(&cnt[dst[e]], 1.0f);
}

// agg[dst[e]] += h[src[e]] ; one thread per (edge, 16-float chunk)
__global__ void scatter_kernel(const float* __restrict__ h,
                               const int* __restrict__ src,
                               const int* __restrict__ dst,
                               float* __restrict__ agg) {
  int t = blockIdx.x * blockDim.x + threadIdx.x;
  if (t >= N_EDGES * 4) return;
  int e = t >> 2;
  int c = (t & 3) << 4;
  int s = src[e];
  int d = dst[e];
  const float4* hp = (const float4*)(h + s * HID + c);
  float*        ap = agg + d * HID + c;
  float4 v0 = hp[0], v1 = hp[1], v2 = hp[2], v3 = hp[3];
  atomicAdd(ap + 0,  v0.x); atomicAdd(ap + 1,  v0.y);
  atomicAdd(ap + 2,  v0.z); atomicAdd(ap + 3,  v0.w);
  atomicAdd(ap + 4,  v1.x); atomicAdd(ap + 5,  v1.y);
  atomicAdd(ap + 6,  v1.z); atomicAdd(ap + 7,  v1.w);
  atomicAdd(ap + 8,  v2.x); atomicAdd(ap + 9,  v2.y);
  atomicAdd(ap + 10, v2.z); atomicAdd(ap + 11, v2.w);
  atomicAdd(ap + 12, v3.x); atomicAdd(ap + 13, v3.y);
  atomicAdd(ap + 14, v3.z); atomicAdd(ap + 15, v3.w);
}

// ---------------------------------------------------------------- SAGE linear
// hout = relu( (agg/max(cnt,1)) @ Wl + bl + hin @ Wr )
// One wave per 16-node tile; 4 column tiles of 16; K=64 as 2 steps of 32.
// 16 v_wmma_f32_16x16x32_f16 per tile; weights staged in LDS as f16.
__global__ __launch_bounds__(256)
void sage_wmma_kernel(const float* __restrict__ hin,
                      const float* __restrict__ agg,
                      const float* __restrict__ cnt,
                      const float* __restrict__ Wl,
                      const float* __restrict__ bl,
                      const float* __restrict__ Wr,
                      float* __restrict__ hout) {
  __shared__ _Float16 sWl[HID * HID];
  __shared__ _Float16 sWr[HID * HID];
  __shared__ float    sb[HID];

  __builtin_prefetch(Wl, 0, 3);               // global_prefetch_b8
  __builtin_prefetch(Wr, 0, 3);

  for (int i = threadIdx.x; i < HID * HID; i += blockDim.x) {
    sWl[i] = (_Float16)Wl[i];
    sWr[i] = (_Float16)Wr[i];
  }
  if (threadIdx.x < HID) sb[threadIdx.x] = bl[threadIdx.x];
  __syncthreads();

  const int lane = threadIdx.x & 31;
  const int wave = threadIdx.x >> 5;
  const int m    = lane & 15;   // row (A) / column (B,C) within 16-tile
  const int hi   = lane >> 4;   // half-wave select
  const int nTiles = N_NODES / 16;   // 6250, exact

  for (int tile = blockIdx.x * 8 + wave; tile < nTiles; tile += gridDim.x * 8) {
    const int row0 = tile * 16;
    const int row  = row0 + m;
    const float dinv = 1.0f / fmaxf(cnt[row], 1.0f);

    // A fragments (ISA 16-bit A 16x32 layout): lane m holds two contiguous
    // 8-wide K runs: [hi*8, hi*8+8) and [hi*8+16, hi*8+24) per 32-wide k-step.
    v16h aM[2], aH[2];
#pragma unroll
    for (int kt = 0; kt < 2; ++kt) {
      const int c0 = kt * 32 + hi * 8;
      const float* ap = agg + row * HID + c0;
      const float* hp = hin + row * HID + c0;
#pragma unroll
      for (int e = 0; e < 8; ++e) {
        aM[kt][e]     = (_Float16)(ap[e] * dinv);
        aM[kt][e + 8] = (_Float16)(ap[16 + e] * dinv);
        aH[kt][e]     = (_Float16)hp[e];
        aH[kt][e + 8] = (_Float16)hp[16 + e];
      }
    }

#pragma unroll
    for (int nt = 0; nt < 4; ++nt) {
      const int col = nt * 16 + m;
      v8f acc = {};
#pragma unroll
      for (int kt = 0; kt < 2; ++kt) {
        // B fragment: lane holds K = k0..k0+7 and k0+16..k0+23 for column `col`.
        const int k0 = kt * 32 + hi * 8;
        v16h bL, bR;
#pragma unroll
        for (int e = 0; e < 16; ++e) {
          bL[e] = sWl[(k0 + e + (e >= 8 ? 8 : 0)) * HID + col];
          bR[e] = sWr[(k0 + e + (e >= 8 ? 8 : 0)) * HID + col];
        }
        acc = __builtin_amdgcn_wmma_f32_16x16x32_f16(
            false, aM[kt], false, bL, (short)0, acc, false, false);
        acc = __builtin_amdgcn_wmma_f32_16x16x32_f16(
            false, aH[kt], false, bR, (short)0, acc, false, false);
      }
      // C/D layout: VGPR r -> row (r + hi*8), column `col`.
#pragma unroll
      for (int r = 0; r < 8; ++r) {
        const int orow = row0 + r + hi * 8;
        const float v = acc[r] + sb[col];
        hout[orow * HID + col] = fmaxf(v, 0.0f);
      }
    }
  }
}

// gsum[batch[n]] += h[n]; gcnt[batch[n]] += 1
__global__ void pool_kernel(const float* __restrict__ h,
                            const int* __restrict__ batch,
                            float* __restrict__ gsum,
                            float* __restrict__ gcnt) {
  int t = blockIdx.x * blockDim.x + threadIdx.x;
  if (t >= N_NODES * 4) return;
  int node = t >> 2;
  int c    = (t & 3) << 4;
  int g    = batch[node];
  const float* hp = h + node * HID + c;
  float*       gp = gsum + g * HID + c;
#pragma unroll
  for (int j = 0; j < 16; ++j) atomicAdd(gp + j, hp[j]);
  if ((t & 3) == 0) atomicAdd(&gcnt[g], 1.0f);
}

// out[g] = (gsum[g]/max(gcnt,1)) @ W_out + b_out   (2048 x 64 @ 64 x 2: tiny)
__global__ void out_kernel(const float* __restrict__ gsum,
                           const float* __restrict__ gcnt,
                           const float* __restrict__ Wout,
                           const float* __restrict__ bout,
                           float* __restrict__ out) {
  int g = blockIdx.x * blockDim.x + threadIdx.x;
  if (g >= N_GRAPHS) return;
  float dinv = 1.0f / fmaxf(gcnt[g], 1.0f);
  float a0 = bout[0], a1 = bout[1];
#pragma unroll
  for (int k = 0; k < HID; ++k) {
    float p = gsum[g * HID + k] * dinv;
    a0 += p * Wout[k * 2 + 0];
    a1 += p * Wout[k * 2 + 1];
  }
  out[g * 2 + 0] = a0;
  out[g * 2 + 1] = a1;
}

// ---------------------------------------------------------------- launcher
extern "C" void kernel_launch(void* const* d_in, const int* in_sizes, int n_in,
                              void* d_out, int out_size, void* d_ws, size_t ws_size,
                              hipStream_t stream) {
  const int*   x    = (const int*)d_in[0];
  const int*   ei   = (const int*)d_in[1];   // [2, N_EDGES]
  const int*   bat  = (const int*)d_in[2];
  const float* emb  = (const float*)d_in[3];
  const float* W1l  = (const float*)d_in[4];
  const float* b1   = (const float*)d_in[5];
  const float* W1r  = (const float*)d_in[6];
  const float* W2l  = (const float*)d_in[7];
  const float* b2   = (const float*)d_in[8];
  const float* W2r  = (const float*)d_in[9];
  const float* Wout = (const float*)d_in[10];
  const float* bout = (const float*)d_in[11];
  float*       out  = (float*)d_out;

  const int* src = ei;
  const int* dst = ei + N_EDGES;

  // Workspace layout (floats): h0 | h1 | agg | cnt | gsum | gcnt  (~78 MB)
  float* h0   = (float*)d_ws;
  float* h1   = h0  + (size_t)N_NODES * HID;
  float* agg  = h1  + (size_t)N_NODES * HID;
  float* cnt  = agg + (size_t)N_NODES * HID;
  float* gsum = cnt + (size_t)N_NODES;
  float* gcnt = gsum + (size_t)N_GRAPHS * HID;

  const int B = 256;
  const int zeroN = N_NODES * HID + N_NODES + N_GRAPHS * HID + N_GRAPHS;
  const int sageBlocks = (N_NODES / 16 + 7) / 8;   // 8 wave-tiles per block

  // clear agg + cnt + gsum + gcnt (contiguous)
  zero_f32<<<(zeroN + B - 1) / B, B, 0, stream>>>(agg, zeroN);

  embed_kernel<<<(N_NODES * 16 + B - 1) / B, B, 0, stream>>>(x, emb, h0);
  degree_kernel<<<(N_EDGES + B - 1) / B, B, 0, stream>>>(dst, cnt);

  // ---- layer 1
  scatter_kernel<<<(N_EDGES * 4 + B - 1) / B, B, 0, stream>>>(h0, src, dst, agg);
  sage_wmma_kernel<<<sageBlocks, B, 0, stream>>>(h0, agg, cnt, W1l, b1, W1r, h1);

  // ---- layer 2 (reuse agg, write back into h0)
  zero_f32<<<(N_NODES * HID + B - 1) / B, B, 0, stream>>>(agg, N_NODES * HID);
  scatter_kernel<<<(N_EDGES * 4 + B - 1) / B, B, 0, stream>>>(h1, src, dst, agg);
  sage_wmma_kernel<<<sageBlocks, B, 0, stream>>>(h1, agg, cnt, W2l, b2, W2r, h0);

  // ---- pool + head
  pool_kernel<<<(N_NODES * 4 + B - 1) / B, B, 0, stream>>>(h0, bat, gsum, gcnt);
  out_kernel<<<(N_GRAPHS + B - 1) / B, B, 0, stream>>>(gsum, gcnt, Wout, bout, out);
}
